// SlotAttention_81346680586267
// MI455X (gfx1250) — compile-verified, matrix operand order checked by reading, not run on hardware
//
#include <hip/hip_runtime.h>

#define BB 128
#define NN 4096
#define DIN 64
#define DD 64
#define KK 8
#define HH 128
#define NITER 3
#define SPLIT 8                      // workgroups per batch in the streaming pass
#define TOKBLK (NN / SPLIT)          // 512 tokens per streaming workgroup
#define CHUNKS (TOKBLK / 256)        // 2 chunks (8 waves x 32 tokens each)

typedef _Float16 f16;
typedef __attribute__((ext_vector_type(16))) _Float16 v16h;
typedef __attribute__((ext_vector_type(8)))  float    v8f;

// Load one WMMA A/B operand chunk (K=32) from a row-major f16 row.
// 16-bit A 16x32 and B 32x16 lane layouts are symmetric:
//   row/col = lane & 15 ; elements [koff, koff+8) and [koff+16, koff+24),
//   koff = (lane >> 4) * 8.
__device__ __forceinline__ v16h ldrow(const f16* p, int koff) {
  union { int4 q[2]; v16h h; } u;
  u.q[0] = *(const int4*)(p + koff);
  u.q[1] = *(const int4*)(p + koff + 16);
  return u.h;
}

__device__ __forceinline__ v8f wmma16(v16h a, v16h b, v8f c) {
  return __builtin_amdgcn_wmma_f32_16x16x32_f16(false, a, false, b, (short)0, c,
                                                false, false);
}

__device__ __forceinline__ unsigned pk2(f16 a, f16 b) {
  union { f16 h[2]; unsigned u; } p; p.h[0] = a; p.h[1] = b; return p.u;
}

__device__ __forceinline__ float sigf(float x) { return 1.0f / (1.0f + __expf(-x)); }

__device__ __forceinline__ void stage_WT(const float* W, f16 (*WT)[64], int tid) {
  const int col = tid & 63, k0 = (tid >> 6) * 16;
#pragma unroll
  for (int j = 0; j < 16; ++j)
    WT[col][k0 + j] = (f16)W[(size_t)(k0 + j) * DD + col];
}

// slot layernorm: wave w owns slot row w; output f16 (A-operand layout rows 0..7)
__device__ __forceinline__ void ln_slots(const float (*src)[64], f16 (*dst)[64],
                                         const float* g, const float* bb,
                                         int wave, int lane) {
  const float x0 = src[wave][lane], x1 = src[wave][lane + 32];
  float s = x0 + x1, sq = x0 * x0 + x1 * x1;
#pragma unroll
  for (int off = 16; off; off >>= 1) {
    s += __shfl_xor(s, off, 32); sq += __shfl_xor(sq, off, 32);
  }
  const float mean = s * (1.f / 64.f);
  const float inv  = rsqrtf(sq * (1.f / 64.f) - mean * mean + 1e-5f);
  dst[wave][lane]      = (f16)((x0 - mean) * inv * g[lane]      + bb[lane]);
  dst[wave][lane + 32] = (f16)((x1 - mean) * inv * g[lane + 32] + bb[lane + 32]);
}

// q = sln @ Wq scaled by D^-0.5, written to global qbuf[b] ([16][64] f16,
// rows 8..15 zero because sln pad rows are zero). Waves 0..3 only (uniform).
__device__ __forceinline__ void compute_q(f16 (*sln)[64], f16 (*WqT)[64],
                                          f16* qdst, int wave, int lane) {
  const int lr = lane & 15, koff = (lane >> 4) * 8;
  if (wave < 4) {
    const int dt = wave;
    const v16h a0 = ldrow(&sln[lr][0], koff), a1 = ldrow(&sln[lr][32], koff);
    v8f acc = {};
    acc = wmma16(a0, ldrow(&WqT[dt*16 + lr][0],  koff), acc);
    acc = wmma16(a1, ldrow(&WqT[dt*16 + lr][32], koff), acc);
#pragma unroll
    for (int r = 0; r < 8; ++r)
      qdst[(size_t)((lane < 16 ? r : 8 + r)) * 64 + dt*16 + lr] =
          (f16)(acc[r] * 0.125f);
  }
}

// ---------------------------------------------------------------------------
// Kernel A: layernorm(inputs) -> k (f16, [b][n][d]) and vT (f16, [b][d][n]).
// k+v in f16 = 128 MB total -> stays resident in the 192 MB L2 for all
// three iteration passes; HBM sees inputs once + k/v writes once.
// ---------------------------------------------------------------------------
__global__ void __launch_bounds__(256)
kv_proj_kernel(const float* __restrict__ inputs,
               const float* __restrict__ Wk, const float* __restrict__ Wv,
               const float* __restrict__ lng, const float* __restrict__ lnb,
               f16* __restrict__ kout, f16* __restrict__ vTout)
{
  __shared__ __align__(16) f16 xln[128][64];
  __shared__ __align__(16) f16 WkT[64][64];
  __shared__ __align__(16) f16 WvT[64][64];
  __shared__ __align__(16) f16 stag[8][16][64];

  const int tid = threadIdx.x;
  const int b   = blockIdx.x >> 5;
  const int n0  = (blockIdx.x & 31) * 128;

  { // layernorm -> LDS f16; 2 threads per token
    const int tok = tid >> 1, half = tid & 1;
    const float4* src =
        (const float4*)(inputs + ((size_t)b * NN + n0 + tok) * DIN + half * 32);
    float x[32]; float s = 0.f, sq = 0.f;
#pragma unroll
    for (int j = 0; j < 8; ++j) {
      float4 t = src[j];
      x[4*j+0] = t.x; x[4*j+1] = t.y; x[4*j+2] = t.z; x[4*j+3] = t.w;
    }
#pragma unroll
    for (int j = 0; j < 32; ++j) { s += x[j]; sq += x[j] * x[j]; }
    s  += __shfl_xor(s , 1, 32);
    sq += __shfl_xor(sq, 1, 32);
    const float mean = s * (1.f / 64.f);
    const float inv  = rsqrtf(sq * (1.f / 64.f) - mean * mean + 1e-5f);
#pragma unroll
    for (int j = 0; j < 32; ++j) {
      const int c = half * 32 + j;
      xln[tok][c] = (f16)((x[j] - mean) * inv * lng[c] + lnb[c]);
    }
  }
  stage_WT(Wk, WkT, tid);
  stage_WT(Wv, WvT, tid);
  __syncthreads();

  const int wave = tid >> 5, lane = tid & 31;
  const int lr = lane & 15, koff = (lane >> 4) * 8;
  const int row = wave * 16 + lr;

  const v16h a0 = ldrow(&xln[row][0],  koff);
  const v16h a1 = ldrow(&xln[row][32], koff);

  // k = xln @ Wk
#pragma unroll
  for (int dt = 0; dt < 4; ++dt) {
    v8f acc = {};
    acc = wmma16(a0, ldrow(&WkT[dt*16 + lr][0],  koff), acc);
    acc = wmma16(a1, ldrow(&WkT[dt*16 + lr][32], koff), acc);
#pragma unroll
    for (int r = 0; r < 8; ++r)
      stag[wave][(lane < 16 ? r : 8 + r)][dt*16 + lr] = (f16)acc[r];
  }
  __syncthreads();
  { // coalesced row-major k write (b128 per lane)
    f16* dst = kout + ((size_t)b * NN + n0 + wave * 16 + lr) * DD;
    const int4* srow = (const int4*)&stag[wave][lr][0];
    int4* drow = (int4*)dst;
#pragma unroll
    for (int qi = 0; qi < 4; ++qi) {
      const int q = (lane >> 4) * 4 + qi;
      drow[q] = srow[q];
    }
  }
  __syncthreads();

  // v = xln @ Wv, stored transposed
#pragma unroll
  for (int dt = 0; dt < 4; ++dt) {
    v8f acc = {};
    acc = wmma16(a0, ldrow(&WvT[dt*16 + lr][0],  koff), acc);
    acc = wmma16(a1, ldrow(&WvT[dt*16 + lr][32], koff), acc);
#pragma unroll
    for (int r = 0; r < 8; ++r)
      stag[wave][(lane < 16 ? r : 8 + r)][dt*16 + lr] = (f16)acc[r];
  }
  __syncthreads();
  { // LDS transpose: lane owns 2 d-columns -> b128 stores into vT[d][n]
#pragma unroll
    for (int dd = 0; dd < 2; ++dd) {
      const int d = lane * 2 + dd;
      unsigned wpk[8];
#pragma unroll
      for (int t = 0; t < 8; ++t)
        wpk[t] = pk2(stag[wave][2*t][d], stag[wave][2*t + 1][d]);
      f16* dst = vTout + ((size_t)b * DD + d) * NN + n0 + wave * 16;
      int4* d4 = (int4*)dst;
      d4[0] = make_int4((int)wpk[0], (int)wpk[1], (int)wpk[2], (int)wpk[3]);
      d4[1] = make_int4((int)wpk[4], (int)wpk[5], (int)wpk[6], (int)wpk[7]);
    }
  }
}

// ---------------------------------------------------------------------------
// Init: slots = mu + sigma*noise -> slotsg; q0 -> qbuf; zero upd/den buffers.
// ---------------------------------------------------------------------------
__global__ void __launch_bounds__(256)
slots_init_kernel(const float* __restrict__ noise, const float* __restrict__ mu,
                  const float* __restrict__ sigma, const float* __restrict__ Wq,
                  const float* __restrict__ lnsg, const float* __restrict__ lnsb,
                  float* __restrict__ slotsg, f16* __restrict__ qbuf,
                  float* __restrict__ updg, float* __restrict__ deng)
{
  __shared__ __align__(16) f16 WqT[64][64];
  __shared__ __align__(16) f16 sln[16][64];
  __shared__ float slots[8][64];
  const int tid = threadIdx.x, b = blockIdx.x;
  const int wave = tid >> 5, lane = tid & 31;

  stage_WT(Wq, WqT, tid);
#pragma unroll
  for (int e = 0; e < 2; ++e) {
    const int idx = tid * 2 + e, i = idx >> 6, d = idx & 63;
    const float s = mu[d] + sigma[d] * noise[(size_t)b * KK * DD + idx];
    slots[i][d] = s;
    slotsg[(size_t)b * KK * DD + idx] = s;
    sln[8 + i][d] = (f16)0.f;
    updg[(size_t)b * KK * DD + idx] = 0.f;
  }
  if (tid < 8) deng[(size_t)b * KK + tid] = 0.f;
  __syncthreads();
  ln_slots(slots, sln, lnsg, lnsb, wave, lane);
  __syncthreads();
  compute_q(sln, WqT, qbuf + (size_t)b * 16 * 64, wave, lane);
}

// ---------------------------------------------------------------------------
// Streaming pass: grid = B*SPLIT workgroups (1024 total) so the whole chip
// participates; each block streams 512 tokens of k/vT out of L2.
// dots(WMMA) -> 8-slot lane-local softmax -> updates(WMMA); LDS reduction,
// then global_atomic_add_f32 into per-batch numerator/denominator.
// ---------------------------------------------------------------------------
__global__ void __launch_bounds__(256)
attn_stream_kernel(const f16* __restrict__ kbuf, const f16* __restrict__ vTbuf,
                   const f16* __restrict__ qbuf,
                   float* __restrict__ updg, float* __restrict__ deng)
{
  __shared__ __align__(16) f16 attnStag[8][16][32];
  __shared__ float upd[8][64];
  __shared__ float den[8];

  const int tid  = threadIdx.x;
  const int b    = blockIdx.x >> 3;       // SPLIT = 8
  const int sp   = blockIdx.x & 7;
  const int wave = tid >> 5, lane = tid & 31;
  const int lr = lane & 15, koff = (lane >> 4) * 8;

#pragma unroll
  for (int e = 0; e < 2; ++e) {
    const int idx = tid * 2 + e;
    upd[idx >> 6][idx & 63] = 0.f;
  }
  if (tid < 8) den[tid] = 0.f;
  __syncthreads();

  const f16* kb = kbuf  + (size_t)b * NN * DD;
  const f16* vb = vTbuf + (size_t)b * DD * NN;
  const f16* qb = qbuf  + (size_t)b * 16 * 64;

  const v16h aq0 = ldrow(qb + lr * 64,      koff);
  const v16h aq1 = ldrow(qb + lr * 64 + 32, koff);
  v8f uc0 = {}, uc1 = {}, uc2 = {}, uc3 = {};
  float dl[8] = {0.f,0.f,0.f,0.f,0.f,0.f,0.f,0.f};

#pragma unroll
  for (int ch = 0; ch < CHUNKS; ++ch) {
    const int nb = sp * TOKBLK + ch * 256 + wave * 32;
    if (ch + 1 < CHUNKS) { // prefetch next chunk (L2 -> near caches)
      const int nb2 = nb + 256;
      __builtin_prefetch(kb + (size_t)(nb2 + lr) * DD, 0, 3);
      __builtin_prefetch(kb + (size_t)(nb2 + 16 + lr) * DD, 0, 3);
#pragma unroll
      for (int dt = 0; dt < 4; ++dt)
        __builtin_prefetch(vb + (size_t)(dt*16 + lr) * NN + nb2, 0, 3);
    }
#pragma unroll
    for (int sub = 0; sub < 2; ++sub) {
      const int n0 = nb + sub * 16;
      const f16* krow = kb + (size_t)(n0 + lr) * DD;
      v8f dc = {};
      dc = wmma16(aq0, ldrow(krow,      koff), dc);
      dc = wmma16(aq1, ldrow(krow + 32, koff), dc);
      // softmax over the 8 slots: one token per lane (lanes 0..15 valid)
      float m = dc[0];
#pragma unroll
      for (int r = 1; r < 8; ++r) m = fmaxf(m, dc[r]);
      float e[8], es = 0.f;
#pragma unroll
      for (int r = 0; r < 8; ++r) { e[r] = __expf(dc[r] - m); es += e[r]; }
      const float isum = 1.f / es;
#pragma unroll
      for (int r = 0; r < 8; ++r) {
        const float at = e[r] * isum + 1e-8f;
        dl[r] += at;
        attnStag[wave][(lane < 16 ? r : 8 + r)][sub * 16 + lr] = (f16)at;
      }
    }
    // attn (A layout, K=32 tokens) x vT tiles
    const v16h aat = ldrow(&attnStag[wave][lr][0], koff);
    uc0 = wmma16(aat, ldrow(vb + (size_t)( 0 + lr) * NN + nb, koff), uc0);
    uc1 = wmma16(aat, ldrow(vb + (size_t)(16 + lr) * NN + nb, koff), uc1);
    uc2 = wmma16(aat, ldrow(vb + (size_t)(32 + lr) * NN + nb, koff), uc2);
    uc3 = wmma16(aat, ldrow(vb + (size_t)(48 + lr) * NN + nb, koff), uc3);
  }
  if (lane < 16) { // cross-wave reduction in LDS (ds_add_f32)
#pragma unroll
    for (int r = 0; r < 8; ++r) {
      atomicAdd(&den[r], dl[r]);
      atomicAdd(&upd[r][ 0 + lane], uc0[r]);
      atomicAdd(&upd[r][16 + lane], uc1[r]);
      atomicAdd(&upd[r][32 + lane], uc2[r]);
      atomicAdd(&upd[r][48 + lane], uc3[r]);
    }
  }
  __syncthreads();
  // one compact set of global atomics per workgroup
#pragma unroll
  for (int e = 0; e < 2; ++e) {
    const int idx = tid * 2 + e;
    atomicAdd(&updg[(size_t)b * KK * DD + idx], upd[idx >> 6][idx & 63]);
  }
  if (tid < 8) atomicAdd(&deng[(size_t)b * KK + tid], den[tid]);
}

// ---------------------------------------------------------------------------
// Per-batch tail: normalize updates, GRU, ln_ff, MLP residual; then either
// emit next iteration's q (and re-zero accumulators) or write the output.
// ---------------------------------------------------------------------------
__global__ void __launch_bounds__(256)
gru_mlp_kernel(const float* __restrict__ Wq,
               const float* __restrict__ W_ih, const float* __restrict__ W_hh,
               const float* __restrict__ b_ih, const float* __restrict__ b_hh,
               const float* __restrict__ W1, const float* __restrict__ b1,
               const float* __restrict__ W2, const float* __restrict__ b2,
               const float* __restrict__ lnsg, const float* __restrict__ lnsb,
               const float* __restrict__ lnfg, const float* __restrict__ lnfb,
               float* __restrict__ slotsg, f16* __restrict__ qbuf,
               float* __restrict__ updg, float* __restrict__ deng,
               float* __restrict__ out, int last)
{
  __shared__ __align__(16) f16 WqT[64][64];
  __shared__ __align__(16) f16 sln[16][64];
  __shared__ float slots[8][64];
  __shared__ float slots_prev[8][64];
  __shared__ float upd[8][64];
  __shared__ float gxb[8][192];
  __shared__ float ghb[8][192];
  __shared__ float hid[8][128];
  __shared__ float den[8];

  const int tid = threadIdx.x, b = blockIdx.x;
  const int wave = tid >> 5, lane = tid & 31;

  stage_WT(Wq, WqT, tid);
#pragma unroll
  for (int e = 0; e < 2; ++e) {
    const int idx = tid * 2 + e, i = idx >> 6, d = idx & 63;
    upd[i][d]        = updg[(size_t)b * KK * DD + idx];
    slots_prev[i][d] = slotsg[(size_t)b * KK * DD + idx];
    sln[8 + i][d]    = (f16)0.f;
  }
  if (tid < 8) den[tid] = deng[(size_t)b * KK + tid];
  __syncthreads();
#pragma unroll
  for (int e = 0; e < 2; ++e) { // normalize; re-zero accumulators for next pass
    const int idx = tid * 2 + e, i = idx >> 6, d = idx & 63;
    upd[i][d] = upd[i][d] / den[i];
    updg[(size_t)b * KK * DD + idx] = 0.f;
  }
  if (tid < 8) deng[(size_t)b * KK + tid] = 0.f;
  __syncthreads();

  // GRU pre-activations
#pragma unroll
  for (int o = 0; o < 6; ++o) {
    const int idx = tid * 6 + o;              // 8 x 192
    const int i = idx / 192, j = idx % 192;
    const float* wi = W_ih + (size_t)j * DD;
    const float* wh = W_hh + (size_t)j * DD;
    float ax = b_ih[j], ah = b_hh[j];
    for (int d = 0; d < DD; ++d) {
      ax += upd[i][d] * wi[d];
      ah += slots_prev[i][d] * wh[d];
    }
    gxb[i][j] = ax; ghb[i][j] = ah;
  }
  __syncthreads();
#pragma unroll
  for (int e = 0; e < 2; ++e) { // gates
    const int idx = tid * 2 + e, i = idx >> 6, d = idx & 63;
    const float r = sigf(gxb[i][d]        + ghb[i][d]);
    const float z = sigf(gxb[i][64 + d]   + ghb[i][64 + d]);
    const float n = tanhf(gxb[i][128 + d] + r * ghb[i][128 + d]);
    slots[i][d] = (1.f - z) * n + z * slots_prev[i][d];
  }
  __syncthreads();
  { // ln_ff in place
    const float x0 = slots[wave][lane], x1 = slots[wave][lane + 32];
    float s = x0 + x1, sq = x0 * x0 + x1 * x1;
#pragma unroll
    for (int off = 16; off; off >>= 1) {
      s += __shfl_xor(s, off, 32); sq += __shfl_xor(sq, off, 32);
    }
    const float mean = s * (1.f / 64.f);
    const float inv  = rsqrtf(sq * (1.f / 64.f) - mean * mean + 1e-5f);
    slots[wave][lane]      = (x0 - mean) * inv * lnfg[lane]      + lnfb[lane];
    slots[wave][lane + 32] = (x1 - mean) * inv * lnfg[lane + 32] + lnfb[lane + 32];
  }
  __syncthreads();
#pragma unroll
  for (int o = 0; o < 4; ++o) { // MLP hidden
    const int idx = tid * 4 + o, i = idx >> 7, hj = idx & 127;
    float a = b1[hj];
    for (int d = 0; d < DD; ++d) a += slots[i][d] * W1[(size_t)d * HH + hj];
    hid[i][hj] = fmaxf(a, 0.f);
  }
  __syncthreads();
#pragma unroll
  for (int e = 0; e < 2; ++e) { // MLP out + residual
    const int idx = tid * 2 + e, i = idx >> 6, d = idx & 63;
    float a = b2[d];
    for (int h = 0; h < HH; ++h) a += hid[i][h] * W2[(size_t)h * DD + d];
    slots[i][d] += a;
  }
  __syncthreads();
#pragma unroll
  for (int e = 0; e < 2; ++e) {
    const int idx = tid * 2 + e, i = idx >> 6, d = idx & 63;
    slotsg[(size_t)b * KK * DD + idx] = slots[i][d];
    if (last) out[(size_t)b * KK * DD + idx] = slots[i][d];
  }
  if (!last) { // next iteration's q
    __syncthreads();
    ln_slots(slots, sln, lnsg, lnsb, wave, lane);
    __syncthreads();
    compute_q(sln, WqT, qbuf + (size_t)b * 16 * 64, wave, lane);
  }
}

extern "C" void kernel_launch(void* const* d_in, const int* in_sizes, int n_in,
                              void* d_out, int out_size, void* d_ws, size_t ws_size,
                              hipStream_t stream)
{
  const float* inputs = (const float*)d_in[0];
  const float* noise  = (const float*)d_in[1];
  const float* mu     = (const float*)d_in[2];
  const float* sigma  = (const float*)d_in[3];
  const float* Wq     = (const float*)d_in[4];
  const float* Wk     = (const float*)d_in[5];
  const float* Wv     = (const float*)d_in[6];
  const float* W_ih   = (const float*)d_in[7];
  const float* W_hh   = (const float*)d_in[8];
  const float* b_ih   = (const float*)d_in[9];
  const float* b_hh   = (const float*)d_in[10];
  const float* W1     = (const float*)d_in[11];
  const float* b1     = (const float*)d_in[12];
  const float* W2     = (const float*)d_in[13];
  const float* b2     = (const float*)d_in[14];
  const float* ling   = (const float*)d_in[15];
  const float* linb   = (const float*)d_in[16];
  const float* lnsg   = (const float*)d_in[17];
  const float* lnsb   = (const float*)d_in[18];
  const float* lnfg   = (const float*)d_in[19];
  const float* lnfb   = (const float*)d_in[20];

  char* base = (char*)d_ws;
  f16*   kbuf   = (f16*)base;                                   // 64 MB
  f16*   vT     = (f16*)(base + (size_t)BB*NN*DD*2);            // 64 MB
  char*  p      = base + (size_t)2*BB*NN*DD*2;
  f16*   qbuf   = (f16*)p;                 p += (size_t)BB*16*64*2;   // 256 KB
  float* updg   = (float*)p;               p += (size_t)BB*KK*DD*4;   // 256 KB
  float* deng   = (float*)p;               p += (size_t)BB*KK*4;      //   4 KB
  float* slotsg = (float*)p;                                          // 256 KB

  kv_proj_kernel<<<BB * (NN / 128), 256, 0, stream>>>(inputs, Wk, Wv, ling, linb,
                                                      kbuf, vT);
  slots_init_kernel<<<BB, 256, 0, stream>>>(noise, mu, sigma, Wq, lnsg, lnsb,
                                            slotsg, qbuf, updg, deng);
  for (int it = 0; it < NITER; ++it) {
    attn_stream_kernel<<<BB * SPLIT, 256, 0, stream>>>(kbuf, vT, qbuf, updg, deng);
    gru_mlp_kernel<<<BB, 256, 0, stream>>>(Wq, W_ih, W_hh, b_ih, b_hh,
                                           W1, b1, W2, b2, lnsg, lnsb, lnfg, lnfb,
                                           slotsg, qbuf, updg, deng,
                                           (float*)d_out, it == NITER - 1 ? 1 : 0);
  }
}
